// StyleLoss_26293789786555
// MI455X (gfx1250) — matrix operand order repaired and checked by reading, not compiled, not run on hardware
//
#include <hip/hip_runtime.h>
#include <stdint.h>
#include <math.h>

// ---------------- problem constants ----------------
#define CCH   512
#define HH    192
#define WW    192
#define AROWS 4608              // C*9
#define LCOLS 4096              // (H/3)*(W/3)
#define GROWB (LCOLS * 2)       // bytes per packed f16 row (8192)

// GEMM tiling
#define BM 128
#define BN 128
#define BK 32
#define KITER (LCOLS / BK)      // 128

// LDS staging: 4 slabs (styHi, styLo, imgHi, imgLo), 128 rows x 32 K f16,
// rows padded 64B -> 80B (16B aligned, spreads banks), double buffered.
#define ROWB  80
#define SLAB  (BM * ROWB)       // 10240
#define HSLB  (64 * ROWB)       // 5120 : rows 64..127 within a slab
#define SLAB4 (4 * SLAB)        // 40960
#define LDS_BYTES (2 * SLAB4)   // 81920

typedef __attribute__((ext_vector_type(8)))  _Float16 v8h;
typedef __attribute__((ext_vector_type(16))) _Float16 v16h;
typedef __attribute__((ext_vector_type(8)))  float    v8f;

// ---------------- helpers ----------------
__device__ __forceinline__ uint32_t f2ord(float f) {
    uint32_t u = __float_as_uint(f);
    return (u & 0x80000000u) ? ~u : (u | 0x80000000u);
}

// Async global->LDS copy, GVS addressing: mem = SGPR64 + VGPR32, ASYNCcnt-tracked.
__device__ __forceinline__ void async_ld_b128(uint32_t lds_off, uint32_t voff,
                                              const char* sbase) {
    asm volatile("global_load_async_to_lds_b128 %0, %1, %2"
                 :: "v"(lds_off), "v"(voff), "s"(sbase) : "memory");
}

__device__ __forceinline__ v8f wmma_f16(v16h a, v16h b, v8f c) {
    return __builtin_amdgcn_wmma_f32_16x16x32_f16(false, a, false, b,
                                                  (short)0, c, false, false);
}

// A operand (16x32 f16, M x K): lane<16 -> row=lane, halves K0..7 | K16..23
//                               lane>=16 -> row=lane-16, K8..15 | K24..31
__device__ __forceinline__ v16h load_a(const char* slab, int rowBase, int lane) {
    int row = rowBase + (lane & 15);
    int o0  = (lane & 16) ? 16 : 0;
    const char* p = slab + row * ROWB + o0;
    v8h x0 = *(const v8h*)(p);
    v8h x1 = *(const v8h*)(p + 32);
    return __builtin_shufflevector(x0, x1, 0,1,2,3,4,5,6,7,8,9,10,11,12,13,14,15);
}

// B operand (32x16 f16, K x N), column n = packed row n (K-contiguous):
// lane<16 -> n=lane, K0..7 | K16..23 ; lane>=16 -> n=lane-16, K8..15 | K24..31
__device__ __forceinline__ v16h load_b(const char* slab, int rowBase, int lane) {
    int row = rowBase + (lane & 15);
    int o0  = (lane & 16) ? 16 : 0;
    const char* p = slab + row * ROWB + o0;
    v8h x0 = *(const v8h*)(p);
    v8h x1 = *(const v8h*)(p + 32);
    return __builtin_shufflevector(x0, x1, 0,1,2,3,4,5,6,7,8,9,10,11,12,13,14,15);
}

// ---------------- kernel 1: unfold + f32->(f16 hi,lo) pack + row norms ----------------
__global__ void pack_norm_kernel(const float* __restrict__ model,
                                 const float* __restrict__ style,
                                 _Float16* __restrict__ imgHi, _Float16* __restrict__ imgLo,
                                 _Float16* __restrict__ styHi, _Float16* __restrict__ styLo,
                                 float* __restrict__ imgInv, float* __restrict__ styInv,
                                 unsigned long long* __restrict__ best) {
    int a   = blockIdx.x;             // unfolded row: a = c*9 + kh*3 + kw
    int tid = threadIdx.x;
    int c = a / 9, r9 = a % 9, kh = r9 / 3, kw = r9 % 3;
    const int HWp = HH * WW;
    float sm = 0.0f, ss = 0.0f;
    size_t rowOff = (size_t)a * LCOLS;
    for (int l = tid; l < LCOLS; l += 256) {
        int ph = l >> 6, pw = l & 63;
        int src = c * HWp + (ph * 3 + kh) * WW + (pw * 3 + kw);
        float xm = model[src], xs = style[src];
        _Float16 hm = (_Float16)xm; _Float16 lm = (_Float16)(xm - (float)hm);
        _Float16 hs = (_Float16)xs; _Float16 ls = (_Float16)(xs - (float)hs);
        imgHi[rowOff + l] = hm; imgLo[rowOff + l] = lm;
        styHi[rowOff + l] = hs; styLo[rowOff + l] = ls;
        sm += xm * xm; ss += xs * xs;
    }
    __shared__ float rm[256], rs[256];
    rm[tid] = sm; rs[tid] = ss;
    __syncthreads();
    for (int s = 128; s > 0; s >>= 1) {
        if (tid < s) { rm[tid] += rm[tid + s]; rs[tid] += rs[tid + s]; }
        __syncthreads();
    }
    if (tid == 0) {
        imgInv[a] = 1.0f / sqrtf(rm[0]);
        styInv[a] = 1.0f / sqrtf(rs[0]);
        best[a]   = 0ull;   // re-init every call (deterministic)
    }
}

// ---------------- kernel 2: WMMA GEMM (split-f16) + fused column argmax ----------------
// Stage one BK=32 slab quad into LDS buffer bufOff (compile-time const).
// Per thread: 8 async b128 copies; all addressing is loop-invariant except the
// wave-uniform SGPR bases (advanced by 64 B per slab by the caller).
__device__ __forceinline__ void stage_slab(uint32_t bufOff, uint32_t ldsT,
                                           uint32_t voffS0, uint32_t voffS1,
                                           uint32_t voffI0, uint32_t voffI1,
                                           const char* pSh, const char* pSl,
                                           const char* pIh, const char* pIl) {
    async_ld_b128(bufOff + 0 * SLAB        + ldsT, voffS0, pSh);
    async_ld_b128(bufOff + 0 * SLAB + HSLB + ldsT, voffS1, pSh);
    async_ld_b128(bufOff + 1 * SLAB        + ldsT, voffS0, pSl);
    async_ld_b128(bufOff + 1 * SLAB + HSLB + ldsT, voffS1, pSl);
    async_ld_b128(bufOff + 2 * SLAB        + ldsT, voffI0, pIh);
    async_ld_b128(bufOff + 2 * SLAB + HSLB + ldsT, voffI1, pIh);
    async_ld_b128(bufOff + 3 * SLAB        + ldsT, voffI0, pIl);
    async_ld_b128(bufOff + 3 * SLAB + HSLB + ldsT, voffI1, pIl);
}

// One BK=32 step: 24 WMMAs, A tiles software-pipelined ahead of use.
__device__ __forceinline__ void compute_slab(const char* lds, uint32_t bufOff,
                                             int wm, int wn, int lane,
                                             v8f (&acc)[4][2]) {
    const char* sAh = lds + bufOff;
    const char* sAl = sAh + SLAB;
    const char* sBh = sAh + 2 * SLAB;
    const char* sBl = sAh + 3 * SLAB;
    v16h bh0 = load_b(sBh, wn, lane);
    v16h bh1 = load_b(sBh, wn + 16, lane);
    v16h bl0 = load_b(sBl, wn, lane);
    v16h bl1 = load_b(sBl, wn + 16, lane);
    v16h ah = load_a(sAh, wm, lane);
    v16h al = load_a(sAl, wm, lane);
    #pragma unroll
    for (int ms = 0; ms < 4; ++ms) {
        v16h ahn = ah, aln = al;
        if (ms < 3) {                               // prefetch next A pair
            ahn = load_a(sAh, wm + 16 * (ms + 1), lane);
            aln = load_a(sAl, wm + 16 * (ms + 1), lane);
        }
        acc[ms][0] = wmma_f16(ah, bh0, acc[ms][0]);
        acc[ms][0] = wmma_f16(ah, bl0, acc[ms][0]);
        acc[ms][0] = wmma_f16(al, bh0, acc[ms][0]);
        acc[ms][1] = wmma_f16(ah, bh1, acc[ms][1]);
        acc[ms][1] = wmma_f16(ah, bl1, acc[ms][1]);
        acc[ms][1] = wmma_f16(al, bh1, acc[ms][1]);
        ah = ahn; al = aln;
    }
}

__global__ void __launch_bounds__(256)
gemm_argmax_kernel(const char* __restrict__ styHi, const char* __restrict__ styLo,
                   const char* __restrict__ imgHi, const char* __restrict__ imgLo,
                   const float* __restrict__ imgInv, const float* __restrict__ styInv,
                   unsigned long long* __restrict__ best) {
    extern __shared__ char lds[];   // dynamic: array index == LDS offset for async ops
    const int tid  = threadIdx.x;
    const int lane = tid & 31;
    const int w    = tid >> 5;          // 8 waves
    const int wm   = (w >> 2) * 64;     // wave M offset within block (0,64)
    const int wn   = (w & 3) * 32;      // wave N offset within block (0..96)
    const int m0   = blockIdx.y * BM;   // sty rows (i)
    const int n0   = blockIdx.x * BN;   // img rows (j)

    // Staging addressing (q = 0..7 -> slab q>>1, row (tid>>2) + (q&1)*64, chunk tid&3)
    const int row0  = tid >> 2;
    const int chunk = tid & 3;
    const uint32_t ldsT   = (uint32_t)(row0 * ROWB + chunk * 16);
    const uint32_t voffS0 = (uint32_t)((m0 + row0) * GROWB + chunk * 16);
    const uint32_t voffS1 = voffS0 + 64u * GROWB;
    const uint32_t voffI0 = (uint32_t)((n0 + row0) * GROWB + chunk * 16);
    const uint32_t voffI1 = voffI0 + 64u * GROWB;
    const char* pSh = styHi;    // wave-uniform SGPR bases, advanced +64 B per slab
    const char* pSl = styLo;
    const char* pIh = imgHi;
    const char* pIl = imgLo;

    v8f acc[4][2] = {};                 // 64x32 per wave = 4x2 subtiles of 16x16

    stage_slab(0u, ldsT, voffS0, voffS1, voffI0, voffI1, pSh, pSl, pIh, pIl);
    pSh += 64; pSl += 64; pIh += 64; pIl += 64;

    for (int kk2 = 0; kk2 < KITER / 2; ++kk2) {
        // ---- phase A: compute slab 2*kk2 from buf0, prefetch slab 2*kk2+1 -> buf1
        stage_slab(SLAB4, ldsT, voffS0, voffS1, voffI0, voffI1, pSh, pSl, pIh, pIl);
        pSh += 64; pSl += 64; pIh += 64; pIl += 64;
        asm volatile("s_wait_asynccnt 8" ::: "memory");
        __syncthreads();
        compute_slab(lds, 0u, wm, wn, lane, acc);
        __syncthreads();

        // ---- phase B: compute slab 2*kk2+1 from buf1, prefetch slab 2*kk2+2 -> buf0
        if (kk2 < KITER / 2 - 1) {
            stage_slab(0u, ldsT, voffS0, voffS1, voffI0, voffI1, pSh, pSl, pIh, pIl);
            pSh += 64; pSl += 64; pIh += 64; pIl += 64;
            asm volatile("s_wait_asynccnt 8" ::: "memory");
        } else {
            asm volatile("s_wait_asynccnt 0" ::: "memory");
        }
        __syncthreads();
        compute_slab(lds, SLAB4, wm, wn, lane, acc);
        __syncthreads();
    }

    // Epilogue: S2[i,j] = G[i,j] * imgInv[i] * styInv[j]; column-wise (over i) max.
    const int halfSel = lane >> 4;      // D layout: lane<16 -> M=r, lane>=16 -> M=8+r
    #pragma unroll
    for (int ns = 0; ns < 2; ++ns) {
        int j = n0 + wn + ns * 16 + (lane & 15);
        float sj = styInv[j];
        unsigned long long bst = 0ull;
        #pragma unroll
        for (int ms = 0; ms < 4; ++ms) {
            int ibase = m0 + wm + ms * 16 + halfSel * 8;
            #pragma unroll
            for (int r = 0; r < 8; ++r) {
                int i = ibase + r;
                float v = acc[ms][ns][r] * imgInv[i] * sj;
                unsigned long long cand =
                    ((unsigned long long)f2ord(v) << 32) |
                    (unsigned long long)(0xFFFFFFFFu - (uint32_t)i);   // ties -> lowest i
                bst = (cand > bst) ? cand : bst;
            }
        }
        // lanes l and l+16 hold the same column j: combine halves
        uint32_t blo = (uint32_t)bst, bhi = (uint32_t)(bst >> 32);
        uint32_t olo = (uint32_t)__shfl_xor((int)blo, 16, 32);
        uint32_t ohi = (uint32_t)__shfl_xor((int)bhi, 16, 32);
        unsigned long long oth = ((unsigned long long)ohi << 32) | (unsigned long long)olo;
        bst = (oth > bst) ? oth : bst;
        if (lane < 16) atomicMax(best + j, bst);
    }
}

// ---------------- kernel 3: decode packed (value, ~index) ----------------
__global__ void finalize_kernel(const unsigned long long* __restrict__ best,
                                float* __restrict__ out) {
    int j = blockIdx.x * 256 + threadIdx.x;
    if (j >= AROWS) return;
    unsigned long long p = best[j];
    uint32_t s = (uint32_t)(p >> 32);
    uint32_t u = (s & 0x80000000u) ? (s ^ 0x80000000u) : ~s;
    uint32_t idx = 0xFFFFFFFFu - (uint32_t)p;
    out[j]         = (float)idx;            // nearest (argmax over i)
    out[AROWS + j] = __uint_as_float(u);    // max_sim
}

// ---------------- launch ----------------
extern "C" void kernel_launch(void* const* d_in, const int* in_sizes, int n_in,
                              void* d_out, int out_size, void* d_ws, size_t ws_size,
                              hipStream_t stream) {
    const float* model = (const float*)d_in[0];   // -> img
    const float* style = (const float*)d_in[1];   // -> sty
    char* ws = (char*)d_ws;

    const size_t MATB = (size_t)AROWS * LCOLS * 2;    // 37,748,736 B per f16 matrix
    _Float16* imgHi = (_Float16*)(ws);
    _Float16* imgLo = (_Float16*)(ws + MATB);
    _Float16* styHi = (_Float16*)(ws + 2 * MATB);
    _Float16* styLo = (_Float16*)(ws + 3 * MATB);
    float*    imgInv = (float*)(ws + 4 * MATB);
    float*    styInv = (float*)(ws + 4 * MATB + AROWS * 4);
    unsigned long long* best = (unsigned long long*)(ws + 4 * MATB + 2 * AROWS * 4);

    pack_norm_kernel<<<AROWS, 256, 0, stream>>>(model, style, imgHi, imgLo, styHi, styLo,
                                                imgInv, styInv, best);
    gemm_argmax_kernel<<<dim3(AROWS / BN, AROWS / BM), 256, LDS_BYTES, stream>>>(
        (const char*)styHi, (const char*)styLo, (const char*)imgHi, (const char*)imgLo,
        imgInv, styInv, best);
    finalize_kernel<<<(AROWS + 255) / 256, 256, 0, stream>>>(best, (float*)d_out);
}
